// ClusterCountPredictor_5093831213701
// MI455X (gfx1250) — matrix operand order repaired and checked by reading, not compiled, not run on hardware
//
#include <hip/hip_runtime.h>
#include <hip/hip_bf16.h>
#include <math.h>

typedef __attribute__((ext_vector_type(16))) _Float16 v16h;
typedef __attribute__((ext_vector_type(8)))  float    v8f;

#define BQ   16
#define DQ   256
#define INDIM (3*DQ + 5)   // 773
#define KPAD 800           // 25 * 32
#define NT1  25            // k-tiles in layer 1
#define MINC 3.0f
#define MAXC 50.0f

// ---- workspace layout (dword offsets; deg array is the runtime-sized tail) ----
#define OFF_SUM   0
#define OFF_SQ    (BQ*DQ)
#define OFF_MAXE  (2*BQ*DQ)
#define OFF_CNT   (3*BQ*DQ)         // unsigned: valid-node count per batch
#define OFF_EB    (OFF_CNT + BQ)    // unsigned: directed same-graph edge count per batch
#define OFF_NPG   (OFF_EB  + BQ)    // unsigned: nodes per graph (from batch_vec)
#define OFF_DSUM  (OFF_NPG + BQ)    // float
#define OFF_DSQ   (OFF_DSUM + BQ)   // float
#define OFF_DMAX  (OFF_DSQ + BQ)    // unsigned
#define OFF_W1S   (OFF_DMAX + BQ)   // f16, WMMA-B-swizzled W1: 4*25*32*16 halves
#define W1S_H     (4*NT1*32*16)     // 51200 halves = 25600 dwords
#define OFF_W2S   (OFF_W1S + W1S_H/2)
#define W2S_H     (2*2*32*16)       // 2048 halves = 1024 dwords
#define OFF_DEG   (OFF_W2S + W2S_H/2)   // unsigned, total_nodes entries

// order-preserving float<->uint encode so atomicMax(u32) == float max
__device__ __forceinline__ unsigned fenc(float f) {
    unsigned u = __float_as_uint(f);
    return (u >> 31) ? ~u : (u | 0x80000000u);
}
__device__ __forceinline__ float fdec(unsigned e) {
    unsigned u = (e >> 31) ? (e & 0x7FFFFFFFu) : ~e;
    return __uint_as_float(u);
}

// ---------------- init: zero the workspace ----------------
__global__ void k_init(unsigned* ws, unsigned total) {
    unsigned i = blockIdx.x * blockDim.x + threadIdx.x;
    if (i < total) ws[i] = 0u;
}

// ---------------- fused masked sum / sumsq / max over N ----------------
// grid = (B, 32); block = D threads (one feature column per thread)
__global__ void k_colstats(const float* __restrict__ x, const float* __restrict__ mask,
                           float* __restrict__ ws, int N, int rowsPerChunk) {
    int b = blockIdx.x;
    int n0 = blockIdx.y * rowsPerChunk;
    int n1 = min(n0 + rowsPerChunk, N);
    int d  = threadIdx.x;

    float s = 0.f, ss = 0.f;
    unsigned me = 0u;  // fenc(-inf) lower bound
    const float* xb = x + ((size_t)b * N) * DQ;
    const float* mb = mask + (size_t)b * N;

    for (int n = n0; n < n1; ++n) {
        if (n + 8 < n1)  // CDNA5 global_prefetch of a later row
            __builtin_prefetch(&xb[(size_t)(n + 8) * DQ + d], 0, 0);
        bool valid = mb[n] > -100000000.0f;
        float v = xb[(size_t)n * DQ + d];
        if (valid) {
            s += v; ss += v * v;
            unsigned e = fenc(v);
            me = me > e ? me : e;
        }
    }
    atomicAdd(&ws[OFF_SUM + b * DQ + d], s);
    atomicAdd(&ws[OFF_SQ  + b * DQ + d], ss);
    if (me) atomicMax((unsigned*)ws + OFF_MAXE + b * DQ + d, me);
}

// ---------------- valid-node counts per batch ----------------
__global__ void k_counts(const float* __restrict__ mask, unsigned* __restrict__ ws, int N) {
    int b = blockIdx.x;
    __shared__ unsigned red[256];
    unsigned c = 0;
    for (int n = threadIdx.x; n < N; n += blockDim.x)
        c += (mask[(size_t)b * N + n] > -100000000.0f) ? 1u : 0u;
    red[threadIdx.x] = c;
    __syncthreads();
    for (int off = 128; off > 0; off >>= 1) {
        if (threadIdx.x < off) red[threadIdx.x] += red[threadIdx.x + off];
        __syncthreads();
    }
    if (threadIdx.x == 0) atomicAdd(&ws[OFF_CNT + b], red[0]);
}

// ---------------- pre-swizzle W1/W2 into f16 WMMA B-operand layout ----------------
// B layout (16-bit, 32x16): lane L -> N = L%16, h = (L>=16), element e (0..15) ->
// K = ktile + (e>=8 ? 16 : 0) + 8h + (e&7). 16 consecutive halves per lane = 1 v16h load.
__global__ void k_cvtW(const float* __restrict__ W1, const float* __restrict__ W2,
                       unsigned* __restrict__ ws) {
    int idx = blockIdx.x * blockDim.x + threadIdx.x;
    _Float16* W1sh = (_Float16*)(ws + OFF_W1S);
    _Float16* W2sh = (_Float16*)(ws + OFF_W2S);
    if (idx < W1S_H) {
        int elem = idx & 15, lane = (idx >> 4) & 31, tile = idx >> 9;
        int kt = tile % NT1, nt = tile / NT1;     // tile = nt*NT1 + kt
        int n = nt * 16 + (lane & 15);
        int k = kt * 32 + ((elem & 8) ? 16 : 0) + ((lane & 16) ? 8 : 0) + (elem & 7);
        W1sh[idx] = (k < INDIM) ? (_Float16)W1[k * 64 + n] : (_Float16)0.f;
    } else if (idx < W1S_H + W2S_H) {
        int j = idx - W1S_H;
        int elem = j & 15, lane = (j >> 4) & 31, tile = j >> 9;  // tile = nt*2 + kt
        int kt = tile & 1, nt = tile >> 1;
        int n = nt * 16 + (lane & 15);
        int k = kt * 32 + ((elem & 8) ? 16 : 0) + ((lane & 16) ? 8 : 0) + (elem & 7);
        W2sh[j] = (_Float16)W2[k * 32 + n];
    }
}

// ---------------- edge pass: deg[src] and per-batch edge counts ----------------
#define EPT 8
__global__ void k_edges(const int* __restrict__ ei, const int* __restrict__ bv,
                        unsigned* __restrict__ ws, long E2) {
    __shared__ unsigned ebL[BQ];
    if (threadIdx.x < BQ) ebL[threadIdx.x] = 0u;
    __syncthreads();

    unsigned* deg = ws + OFF_DEG;
    long base = ((long)blockIdx.x * blockDim.x + threadIdx.x) * EPT;
    for (int i = 0; i < EPT; ++i) {
        long e = base + i;
        if (e < E2) {
            int s = ei[e];
            int t = ei[E2 + e];
            int bs = bv[s];
            if (bs == bv[t]) {
                atomicAdd(&deg[s], 1u);
                atomicAdd(&ebL[bs & (BQ - 1)], 1u);   // LDS-privatized counter
            }
        }
    }
    __syncthreads();
    if (threadIdx.x < BQ) atomicAdd(&ws[OFF_EB + threadIdx.x], ebL[threadIdx.x]);
}

// ---------------- per-batch degree stats + nodes-per-graph ----------------
__global__ void k_degstats(const int* __restrict__ bv, float* __restrict__ wsf, int totalNodes) {
    __shared__ float    dsL[BQ], dqL[BQ];
    __shared__ unsigned dmL[BQ], npL[BQ];
    if (threadIdx.x < BQ) { dsL[threadIdx.x] = 0.f; dqL[threadIdx.x] = 0.f;
                            dmL[threadIdx.x] = 0u;  npL[threadIdx.x] = 0u; }
    __syncthreads();

    unsigned* ws = (unsigned*)wsf;
    const unsigned* deg = ws + OFF_DEG;
    int stride = gridDim.x * blockDim.x;
    for (int i = blockIdx.x * blockDim.x + threadIdx.x; i < totalNodes; i += stride) {
        int b = bv[i] & (BQ - 1);
        unsigned du = deg[i];
        float d = (float)du;
        atomicAdd(&dsL[b], d);
        atomicAdd(&dqL[b], d * d);
        atomicMax(&dmL[b], du);
        atomicAdd(&npL[b], 1u);
    }
    __syncthreads();
    if (threadIdx.x < BQ) {
        atomicAdd(&wsf[OFF_DSUM + threadIdx.x], dsL[threadIdx.x]);
        atomicAdd(&wsf[OFF_DSQ  + threadIdx.x], dqL[threadIdx.x]);
        atomicMax(ws + OFF_DMAX + threadIdx.x, dmL[threadIdx.x]);
        atomicAdd(ws + OFF_NPG  + threadIdx.x, npL[threadIdx.x]);
    }
}

// ---------------- finalize: build gf (16 x 773, f16) then WMMA MLP ----------------
__global__ void __launch_bounds__(256)
k_mlp(const float* __restrict__ wsf, const float* __restrict__ b1,
      const float* __restrict__ b2,  const float* __restrict__ W3,
      const float* __restrict__ b3,  float* __restrict__ out) {
    __shared__ _Float16 gfS[BQ][KPAD];      // 25.6 KB
    __shared__ _Float16 h1S[16][64];        // layer-1 activations
    __shared__ float    h2S[16][32];        // layer-2 activations
    __shared__ float    redR[BQ], redMA[BQ], redRT[BQ];

    const unsigned*  wsu  = (const unsigned*)wsf;
    const _Float16*  W1sh = (const _Float16*)(wsu + OFF_W1S);
    const _Float16*  W2sh = (const _Float16*)(wsu + OFF_W2S);

    // ---- assemble gf in f16, zero-padded to KPAD ----
    for (int b = 0; b < BQ; ++b) {
        float cnt = (float)wsu[OFF_CNT + b];
        for (int idx = threadIdx.x; idx < KPAD; idx += blockDim.x) {
            float val = 0.f;
            if (idx < DQ) {                                   // x_mean
                val = wsf[OFF_SUM + b * DQ + idx] / fmaxf(cnt, 1.f);
            } else if (idx < 2 * DQ) {                        // x_max
                int d = idx - DQ;
                val = (cnt > 0.f) ? fdec(wsu[OFF_MAXE + b * DQ + d]) : 0.f;
            } else if (idx < 3 * DQ) {                        // x_std
                int d = idx - 2 * DQ;
                float s  = wsf[OFF_SUM + b * DQ + d];
                float sq = wsf[OFF_SQ  + b * DQ + d];
                float mean = s / fmaxf(cnt, 1.f);
                float var = (sq - cnt * mean * mean) / fmaxf(cnt - 1.f, 1.f);
                val = (cnt > 1.f) ? sqrtf(fmaxf(var, 0.f)) : 0.f;
            } else if (idx < INDIM) {                         // structural[5]
                float npg  = (float)wsu[OFF_NPG + b];
                float Eb   = (float)wsu[OFF_EB  + b];
                float dsum = wsf[OFF_DSUM + b];
                float dsq  = wsf[OFF_DSQ  + b];
                float dmax = (float)wsu[OFF_DMAX + b];
                float num_edges = floorf(Eb * 0.5f);
                float deg_mean = dsum / fmaxf(npg, 1.f);
                float deg_var = (dsq - npg * deg_mean * deg_mean) / fmaxf(npg - 1.f, 1.f);
                float deg_std = sqrtf(fmaxf(deg_var, 0.f));
                float max_edges = cnt * (cnt - 1.f) * 0.5f;
                bool  has = (Eb > 0.f) && (cnt > 1.f);
                int f = idx - 3 * DQ;
                if      (f == 0) val = logf(cnt + 1.f) * 0.2f;
                else if (f == 1) val = has ? num_edges / fmaxf(max_edges, 1.f) : 0.f;
                else if (f == 2) val = has ? deg_mean * 0.1f : 0.f;
                else if (f == 3) val = has ? dmax / fmaxf(cnt, 1.f) : 0.f;
                else             val = (has && npg > 1.f) ? deg_std * 0.1f : 0.f;
            }
            gfS[b][idx] = (_Float16)val;
        }
    }
    __syncthreads();

    // ---- wave 0 runs the MLP with v_wmma_f32_16x16x32_f16 (EXEC all ones here) ----
    if (threadIdx.x < 32) {
        int  lane = threadIdx.x;
        int  M    = lane & 15;
        bool hi   = lane >= 16;

        // Layer 1: (16 x KPAD) @ (KPAD x 64); B tiles are pre-swizzled v16h loads
        for (int nt = 0; nt < 4; ++nt) {
            v8f c = {};
            int n = nt * 16 + (lane & 15);
            for (int t = 0; t < NT1; ++t) {
                int kt = t * 32;
                v16h a;
#pragma unroll
                for (int r = 0; r < 8; ++r) {   // A: 16-bit 16x32 layout (2x ds_load_b128)
                    int ka = kt + ((r & 4) ? 16 : 0) + (hi ? 8 : 0) + (r & 3) * 2;
                    a[2 * r]     = gfS[M][ka];
                    a[2 * r + 1] = gfS[M][ka + 1];
                }
                v16h bm = *reinterpret_cast<const v16h*>(
                    W1sh + ((size_t)(nt * NT1 + t) * 32 + lane) * 16);
                c = __builtin_amdgcn_wmma_f32_16x16x32_f16(false, a, false, bm,
                                                           (short)0, c, false, false);
            }
            float bias = b1[n];
#pragma unroll
            for (int r = 0; r < 8; ++r) {       // C/D layout: M = r + (hi?8:0), N = lane%16
                int m = r + (hi ? 8 : 0);
                h1S[m][n] = (_Float16)fmaxf(c[r] + bias, 0.f);
            }
        }

        // Layer 2: (16 x 64) @ (64 x 32)
        for (int nt = 0; nt < 2; ++nt) {
            v8f c = {};
            int n = nt * 16 + (lane & 15);
            for (int t = 0; t < 2; ++t) {
                int kt = t * 32;
                v16h a;
#pragma unroll
                for (int r = 0; r < 8; ++r) {
                    int ka = kt + ((r & 4) ? 16 : 0) + (hi ? 8 : 0) + (r & 3) * 2;
                    a[2 * r]     = h1S[M][ka];
                    a[2 * r + 1] = h1S[M][ka + 1];
                }
                v16h bm = *reinterpret_cast<const v16h*>(
                    W2sh + ((size_t)(nt * 2 + t) * 32 + lane) * 16);
                c = __builtin_amdgcn_wmma_f32_16x16x32_f16(false, a, false, bm,
                                                           (short)0, c, false, false);
            }
            float bias = b2[n];
#pragma unroll
            for (int r = 0; r < 8; ++r) {
                int m = r + (hi ? 8 : 0);
                h2S[m][n] = fmaxf(c[r] + bias, 0.f);
            }
        }

        // Layer 3 (N=1) + per-batch epilogue on lanes 0..15
        if (lane < BQ) {
            int b = lane;
            float acc = b3[0];
#pragma unroll
            for (int j = 0; j < 32; ++j) acc += h2S[b][j] * W3[j];
            float score = 1.f / (1.f + expf(-acc));

            float cnt = (float)wsu[OFF_CNT + b];
            float safe_nc     = fmaxf(cnt, 1.f);
            float max_allowed = fminf(safe_nc, MAXC);
            float min_allowed = fminf(max_allowed, MINC);
            float ncc = MINC + score * (MAXC - MINC);
            ncc = fmaxf(fminf(ncc, max_allowed), min_allowed);

            redR[b]  = rintf(ncc);          // round-half-even, matches jnp.round
            redMA[b] = max_allowed;
            redRT[b] = ncc / safe_nc;
        }
        if (lane == 0) {
            float mr = 0.f, mm = 3.4e38f, rt = 0.f;
            for (int b = 0; b < BQ; ++b) {
                mr += redR[b];
                mm = fminf(mm, redMA[b]);
                rt += redRT[b];
            }
            mr *= (1.f / BQ);
            rt *= (1.f / BQ);
            int maxb = (int)mm;             // astype(int32) truncation
            int nf   = (int)mr;
            nf = nf < 1 ? 1 : (nf > maxb ? maxb : nf);
            out[0] = (float)nf;
            out[1] = rt;
        }
    }
}

extern "C" void kernel_launch(void* const* d_in, const int* in_sizes, int n_in,
                              void* d_out, int out_size, void* d_ws, size_t ws_size,
                              hipStream_t stream) {
    const float* x    = (const float*)d_in[0];
    const float* mask = (const float*)d_in[1];
    const int*   ei   = (const int*)d_in[3];
    const int*   bv   = (const int*)d_in[4];
    const float* W1   = (const float*)d_in[5];
    const float* b1   = (const float*)d_in[6];
    const float* W2   = (const float*)d_in[7];
    const float* b2   = (const float*)d_in[8];
    const float* W3   = (const float*)d_in[9];
    const float* b3   = (const float*)d_in[10];

    int  totalNodes = in_sizes[4];
    int  N          = totalNodes / BQ;
    long E2         = (long)in_sizes[3] / 2;

    unsigned totDw = (unsigned)(OFF_DEG + totalNodes);
    k_init<<<(totDw + 255) / 256, 256, 0, stream>>>((unsigned*)d_ws, totDw);

    int chunks = 32;
    k_colstats<<<dim3(BQ, chunks), DQ, 0, stream>>>(x, mask, (float*)d_ws, N,
                                                    (N + chunks - 1) / chunks);
    k_counts<<<BQ, 256, 0, stream>>>(mask, (unsigned*)d_ws, N);

    k_cvtW<<<(W1S_H + W2S_H + 255) / 256, 256, 0, stream>>>(W1, W2, (unsigned*)d_ws);

    long perBlock = 256L * EPT;
    int  eblocks  = (int)((E2 + perBlock - 1) / perBlock);
    k_edges<<<eblocks, 256, 0, stream>>>(ei, bv, (unsigned*)d_ws, E2);

    k_degstats<<<256, 256, 0, stream>>>(bv, (float*)d_ws, totalNodes);

    k_mlp<<<1, 256, 0, stream>>>((const float*)d_ws, b1, b2, W3, b3, (float*)d_out);
}
